// SparseKANLinear_45921790329044
// MI455X (gfx1250) — matrix operand order repaired and checked by reading, not compiled, not run on hardware
//
#include <hip/hip_runtime.h>

typedef __attribute__((ext_vector_type(16))) __bf16          v16bf;
typedef __attribute__((ext_vector_type(8)))  __bf16          v8bf;
typedef __attribute__((ext_vector_type(8)))  float           v8f;
typedef __attribute__((ext_vector_type(8)))  unsigned short  ushort8;
typedef __attribute__((ext_vector_type(4)))  int             v4i;

#define IN_F   512
#define OUT_F  512
#define NROWS  16384
#define COEF   8            // spline coeffs per (o,i)
#define KPF    9            // K-values per input feature: 1 silu + 8 bases
#define KTOT   (IN_F * KPF) // 4608
#define BM     128
#define BN     128

// fused-kernel chunking (feature-aligned: 32 features * 9 = 288 K)
#define CI     32
#define KC     (CI * KPF)   // 288
#define LDA    296          // padded K-stride (ushorts)

// two-phase GEMM chunking (flat K, double-buffered)
#define KCG    128          // K per chunk
#define LDG    136          // padded K-stride (68 dwords == 4 mod 64: conflict-free)
#define NCH    (KTOT / KCG) // 36

#define WC_BYTES ((size_t)OUT_F * KTOT * 2)          // 4.5 MB fused weights
#define A_BYTES  ((size_t)NROWS * KTOT * 2)          // 144 MB bf16 activations

// ---------------- CDNA5 async global->LDS copy (validated round 3) ----------------
#if __has_builtin(__builtin_amdgcn_global_load_async_to_lds_b128)
#define ASYNC_OK 1
#else
#define ASYNC_OK 0
#endif

__device__ __forceinline__ void async_cp16(const unsigned short* g, unsigned short* l) {
#if ASYNC_OK
  __builtin_amdgcn_global_load_async_to_lds_b128(
      (__attribute__((address_space(1))) v4i*)g,
      (__attribute__((address_space(3))) v4i*)l, 0, 0);
#else
  *(ushort8*)l = *(const ushort8*)g;
#endif
}

__device__ __forceinline__ void async_wait() {
#if ASYNC_OK
#if __has_builtin(__builtin_amdgcn_s_wait_asynccnt)
  __builtin_amdgcn_s_wait_asynccnt(0);
#else
  asm volatile("s_wait_asynccnt 0x0" ::: "memory");
#endif
#endif
}

// ---------------- helpers ----------------
__device__ __forceinline__ unsigned short f2bf(float f) {
  unsigned int u = __float_as_uint(f);
  u += 0x7FFFu + ((u >> 16) & 1u);           // round-to-nearest-even
  return (unsigned short)(u >> 16);
}

// Cubic B-spline bases on the uniform grid g_j = (j-3)*0.4 - 1, j = 0..11.
__device__ __forceinline__ void bspline8(float x, float* out8) {
  const float h = 0.4f;
  float t[11];
#pragma unroll
  for (int j = 0; j < 11; ++j) {
    float g0 = (float)(j - 3) * h - 1.0f;
    float g1 = (float)(j - 2) * h - 1.0f;
    t[j] = (x >= g0 && x < g1) ? 1.0f : 0.0f;
  }
#pragma unroll
  for (int k = 1; k <= 3; ++k) {
    float inv = 1.0f / ((float)k * h + 1e-8f);
#pragma unroll
    for (int j = 0; j < 11 - k; ++j) {
      float gj  = (float)(j - 3) * h - 1.0f;
      float gk1 = (float)(j + k - 2) * h - 1.0f;   // g[j+k+1]
      t[j] = (x - gj) * inv * t[j] + (gk1 - x) * inv * t[j + 1];
    }
  }
#pragma unroll
  for (int k = 0; k < 8; ++k) out8[k] = t[k];
}

// A 16x32 bf16 fragment: lanes 0-15 K{0-7,16-23}, lanes 16-31 K{8-15,24-31}
__device__ __forceinline__ v16bf load_afrag(const unsigned short* s, int ld, int row,
                                            int kb, int lhalf) {
  const unsigned short* base = &s[row * ld + kb + (lhalf ? 8 : 0)];
  v8bf lo = *(const v8bf*)(base);
  v8bf hi = *(const v8bf*)(base + 16);
  return __builtin_shufflevector(lo, hi, 0, 1, 2, 3, 4, 5, 6, 7,
                                         8, 9, 10, 11, 12, 13, 14, 15);
}
// B 32x16 bf16 fragment: lanes 0-15 K{0-7,16-23}, lanes 16-31 K{8-15,24-31}
__device__ __forceinline__ v16bf load_bfrag(const unsigned short* s, int ld, int row,
                                            int kb, int lhalf) {
  const unsigned short* base = &s[row * ld + kb + (lhalf ? 8 : 0)];
  v8bf lo = *(const v8bf*)(base);
  v8bf hi = *(const v8bf*)(base + 16);
  return __builtin_shufflevector(lo, hi, 0, 1, 2, 3, 4, 5, 6, 7,
                                         8, 9, 10, 11, 12, 13, 14, 15);
}

// ---------------- prep: fused bf16 weights Wc[o, i*9 + {0..8}] ----------------
__global__ __launch_bounds__(256) void prep_wc(const float* __restrict__ bw,
                                               const float* __restrict__ sw,
                                               const float* __restrict__ ss,
                                               const float* __restrict__ mk,
                                               unsigned short* __restrict__ wc) {
  int idx = blockIdx.x * 256 + threadIdx.x;
  if (idx >= OUT_F * IN_F) return;
  int o = idx >> 9;
  int i = idx & (IN_F - 1);
  float m = mk[idx];
  float b = bw[idx] * m;
  float s = ss[idx] * m;
  unsigned short* dst = &wc[(size_t)o * KTOT + i * KPF];
  dst[0] = f2bf(b);
#pragma unroll
  for (int k = 0; k < COEF; ++k)
    dst[1 + k] = f2bf(sw[(size_t)idx * COEF + k] * s);
}

// ---------------- phase 1: build bf16 activation matrix A[n, i*9 + {0..8}] ----------------
__global__ __launch_bounds__(256) void build_a(const float* __restrict__ X,
                                               unsigned short* __restrict__ Aw) {
  int idx = blockIdx.x * 256 + threadIdx.x;     // one thread per (n, i)
  int n = idx >> 9;
  int i = idx & (IN_F - 1);
  float xv = X[idx];                            // coalesced
  float sv = xv / (1.0f + __expf(-xv));         // SiLU
  float bb[8];
  bspline8(xv, bb);
  unsigned short* dst = &Aw[(size_t)n * KTOT + i * KPF];
  dst[0] = f2bf(sv);
#pragma unroll
  for (int k = 0; k < 8; ++k) dst[1 + k] = f2bf(bb[k]);
}

// ------- phase 2: double-buffered async-pipelined bf16 WMMA GEMM: Out = A @ Wc^T -------
__global__ __launch_bounds__(256) void kan_gemm(const unsigned short* __restrict__ Aw,
                                                const unsigned short* __restrict__ Wc,
                                                float* __restrict__ Out) {
  __shared__ unsigned short sA[2][BM * LDG];    // 2 x 34 KB
  __shared__ unsigned short sB[2][BN * LDG];    // 2 x 34 KB   (total 136 KB)

  const int tid   = threadIdx.x;
  const int lane  = tid & 31;
  const int wid   = tid >> 5;
  const int wm    = wid >> 2;      // 0..1
  const int wn    = wid & 3;       // 0..3
  const int nBase = blockIdx.x * BM;
  const int oBase = blockIdx.y * BN;
  const int lhalf = (lane >= 16) ? 1 : 0;
  const int l15   = lane & 15;

  // this thread's copy slice: 8 x 16B per matrix per chunk
  const int cr = tid >> 4;               // row pair base: 16 threads cover one row's 128 K
  const int ck = (tid & 15) * 8;         // k8 offset within row

  v8f acc[4][2];
#pragma unroll
  for (int a = 0; a < 4; ++a)
#pragma unroll
    for (int b = 0; b < 2; ++b)
#pragma unroll
      for (int e = 0; e < 8; ++e) acc[a][b][e] = 0.0f;

  // prologue: issue chunk 0 into buffer 0
#pragma unroll
  for (int p = 0; p < 8; ++p) {
    int r = cr + p * 16;                 // 16 row-groups x 8 = 128 rows
    async_cp16(&Aw[(size_t)(nBase + r) * KTOT + ck], &sA[0][r * LDG + ck]);
    async_cp16(&Wc[(size_t)(oBase + r) * KTOT + ck], &sB[0][r * LDG + ck]);
  }

  for (int ci = 0; ci < NCH; ++ci) {
    const int buf = ci & 1;
    async_wait();                        // own loads for chunk ci done
    __syncthreads();                     // => all waves' loads for chunk ci done

    if (ci + 1 < NCH) {                  // prefetch chunk ci+1 into other buffer
      const int nb = buf ^ 1;
      const size_t gk = (size_t)(ci + 1) * KCG + ck;
#pragma unroll
      for (int p = 0; p < 8; ++p) {
        int r = cr + p * 16;
        async_cp16(&Aw[(size_t)(nBase + r) * KTOT + gk], &sA[nb][r * LDG + ck]);
        async_cp16(&Wc[(size_t)(oBase + r) * KTOT + gk], &sB[nb][r * LDG + ck]);
      }
    }

#pragma unroll
    for (int ks = 0; ks < KCG / 32; ++ks) {       // 4 k-steps
      const int kb = ks * 32;
      v16bf afrag[4];
      v16bf bfrag[2];
#pragma unroll
      for (int tm = 0; tm < 4; ++tm)
        afrag[tm] = load_afrag(sA[buf], LDG, wm * 64 + tm * 16 + l15, kb, lhalf);
#pragma unroll
      for (int tn = 0; tn < 2; ++tn)
        bfrag[tn] = load_bfrag(sB[buf], LDG, wn * 32 + tn * 16 + l15, kb, lhalf);
#pragma unroll
      for (int tm = 0; tm < 4; ++tm)
#pragma unroll
        for (int tn = 0; tn < 2; ++tn)
          acc[tm][tn] = __builtin_amdgcn_wmma_f32_16x16x32_bf16(
              false, afrag[tm], false, bfrag[tn], (short)0, acc[tm][tn],
              false, false);
    }
    __syncthreads();                     // reads of buf done before it is refilled
  }

#pragma unroll
  for (int tm = 0; tm < 4; ++tm) {
    int n0 = nBase + wm * 64 + tm * 16 + (lhalf ? 8 : 0);
#pragma unroll
    for (int tn = 0; tn < 2; ++tn) {
      int o0 = oBase + wn * 32 + tn * 16 + l15;
#pragma unroll
      for (int r = 0; r < 8; ++r)
        Out[(size_t)(n0 + r) * OUT_F + o0] = acc[tm][tn][r];
    }
  }
}

// ---------------- fallback: fused activation + GEMM (small workspace) ----------------
__global__ __launch_bounds__(256) void kan_fused(const float* __restrict__ X,
                                                 const unsigned short* __restrict__ Wc,
                                                 float* __restrict__ Out) {
  __shared__ unsigned short sA[BM * LDA];
  __shared__ unsigned short sB[BN * LDA];

  const int tid   = threadIdx.x;
  const int lane  = tid & 31;
  const int wid   = tid >> 5;
  const int wm    = wid >> 2;
  const int wn    = wid & 3;
  const int nBase = blockIdx.x * BM;
  const int oBase = blockIdx.y * BN;
  const int lhalf = (lane >= 16) ? 1 : 0;
  const int l15   = lane & 15;

  v8f acc[4][2];
#pragma unroll
  for (int a = 0; a < 4; ++a)
#pragma unroll
    for (int b = 0; b < 2; ++b)
#pragma unroll
      for (int e = 0; e < 8; ++e) acc[a][b][e] = 0.0f;

  for (int ci = 0; ci < IN_F / CI; ++ci) {
#pragma unroll
    for (int p = 0; p < (BM * CI) / 256; ++p) {       // compute silu + bases
      int idx  = tid + 256 * p;
      int row  = idx >> 5;
      int feat = idx & (CI - 1);
      float xv = X[(size_t)(nBase + row) * IN_F + ci * CI + feat];
      float sv = xv / (1.0f + __expf(-xv));
      float bb[8];
      bspline8(xv, bb);
      unsigned short* dst = &sA[row * LDA + feat * KPF];
      dst[0] = f2bf(sv);
#pragma unroll
      for (int k = 0; k < 8; ++k) dst[1 + k] = f2bf(bb[k]);
    }
#pragma unroll
    for (int p = 0; p < (BN * (KC / 8)) / 256; ++p) { // weights via async copy
      int flat = tid + 256 * p;
      int orow = flat / (KC / 8);
      int k8   = (flat % (KC / 8)) * 8;
      async_cp16(&Wc[(size_t)(oBase + orow) * KTOT + ci * KC + k8], &sB[orow * LDA + k8]);
    }
    async_wait();
    __syncthreads();

#pragma unroll
    for (int ks = 0; ks < KC / 32; ++ks) {
      const int kb = ks * 32;
      v16bf afrag[4];
      v16bf bfrag[2];
#pragma unroll
      for (int tm = 0; tm < 4; ++tm)
        afrag[tm] = load_afrag(sA, LDA, wm * 64 + tm * 16 + l15, kb, lhalf);
#pragma unroll
      for (int tn = 0; tn < 2; ++tn)
        bfrag[tn] = load_bfrag(sB, LDA, wn * 32 + tn * 16 + l15, kb, lhalf);
#pragma unroll
      for (int tm = 0; tm < 4; ++tm)
#pragma unroll
        for (int tn = 0; tn < 2; ++tn)
          acc[tm][tn] = __builtin_amdgcn_wmma_f32_16x16x32_bf16(
              false, afrag[tm], false, bfrag[tn], (short)0, acc[tm][tn],
              false, false);
    }
    __syncthreads();
  }

#pragma unroll
  for (int tm = 0; tm < 4; ++tm) {
    int n0 = nBase + wm * 64 + tm * 16 + (lhalf ? 8 : 0);
#pragma unroll
    for (int tn = 0; tn < 2; ++tn) {
      int o0 = oBase + wn * 32 + tn * 16 + l15;
#pragma unroll
      for (int r = 0; r < 8; ++r)
        Out[(size_t)(n0 + r) * OUT_F + o0] = acc[tm][tn][r];
    }
  }
}

extern "C" void kernel_launch(void* const* d_in, const int* in_sizes, int n_in,
                              void* d_out, int out_size, void* d_ws, size_t ws_size,
                              hipStream_t stream) {
  (void)in_sizes; (void)n_in; (void)out_size;
  const float* x  = (const float*)d_in[0];
  const float* bw = (const float*)d_in[1];
  const float* sw = (const float*)d_in[2];
  const float* ss = (const float*)d_in[3];
  const float* mk = (const float*)d_in[4];
  // d_in[5]: uniform grid, evaluated analytically (g_j = (j-3)*0.4 - 1)
  float* out = (float*)d_out;

  unsigned short* wc = (unsigned short*)d_ws;
  prep_wc<<<(OUT_F * IN_F + 255) / 256, 256, 0, stream>>>(bw, sw, ss, mk, wc);

  dim3 grid(NROWS / BM, OUT_F / BN);              // 128 x 4 blocks

  if (ws_size >= WC_BYTES + A_BYTES) {
    // Two-phase: materialize bf16 activations once (no 4x recompute), then
    // a double-buffered async-pipelined GEMM.
    unsigned short* aw = (unsigned short*)((char*)d_ws + WC_BYTES);
    build_a<<<(NROWS * IN_F) / 256, 256, 0, stream>>>(x, aw);
    kan_gemm<<<grid, 256, 0, stream>>>(aw, wc, out);
  } else {
    kan_fused<<<grid, 256, 0, stream>>>(x, wc, out);
  }
}
